// MessagePassingGNN_73564199845905
// MI455X (gfx1250) — matrix operand order repaired and checked by reading, not compiled, not run on hardware
//
#include <hip/hip_runtime.h>

#define NN 40000
#define NE 640000
#define DHID 128
#define DMSP 256

// LDS tile strides (in elements), padded for bank-conflict-free access.
// XSTR/HSTR multiples of 8 (keep 16B alignment of v8h loads).
#define XSTR 136
#define HSTR 264
#define OSTR 132
#define ROWS 32  // rows per block iteration (two 16-row WMMA subtiles)

typedef __attribute__((ext_vector_type(16))) _Float16 v16h;
typedef __attribute__((ext_vector_type(8)))  _Float16 v8h;
typedef __attribute__((ext_vector_type(8)))  float    v8f;

__device__ __forceinline__ v8f wmma_f32_16x16x32(v16h a, v16h b, v8f c) {
  return __builtin_amdgcn_wmma_f32_16x16x32_f16(false, a, false, b, (short)0, c,
                                                false, false);
}

// ---- B fragment (32x16 f16) from row-major weights W[K][NC] -----------------
// Lane layout: N = lane%16, half = lane/16; element j holds K = koff + half*16 + j
__device__ __forceinline__ v16h load_bfrag(const _Float16* __restrict__ W, int NC,
                                           int koff, int colBase, int lane) {
  int n = lane & 15, half = lane >> 4;
  const _Float16* base = W + (size_t)(koff + half * 16) * NC + colBase + n;
  v16h b;
#pragma unroll
  for (int j = 0; j < 16; ++j) b[j] = base[(size_t)j * NC];
  return b;
}

// ---- A fragment (16x32 f16) from LDS row-major tile (stride mult of 8) ------
// Lane: M = lane%16, half = lane/16.
__device__ __forceinline__ v16h load_afrag(const _Float16* __restrict__ T,
                                           int stride, int koff, int lane) {
  int m = lane & 15, half = lane >> 4;
  const _Float16* r = T + m * stride + koff + half * 16;
  v8h lo = *(const v8h*)(r);
  v8h hi = *(const v8h*)(r + 8);
  return __builtin_shufflevector(lo, hi, 0, 1, 2, 3, 4, 5, 6, 7, 8, 9, 10, 11,
                                 12, 13, 14, 15);
}

// ---- per-wave weight fragments (held in VGPRs for whole kernel) -------------
struct WFrags {
  v16h w1f[2][4];  // wave owns hidden col tiles 2w, 2w+1 ; K = 128 -> 4 slices
  v16h w2f[8];     // wave owns out col tile w          ; K = 256 -> 8 slices
  float bias1[2];
  float bias2;
};

__device__ __forceinline__ void load_wfrags(WFrags& f,
                                            const _Float16* __restrict__ W1,
                                            const float* __restrict__ b1,
                                            const _Float16* __restrict__ W2,
                                            const float* __restrict__ b2,
                                            int wave, int lane) {
  int n = lane & 15;
#pragma unroll
  for (int t = 0; t < 2; ++t) {
    int colBase = (wave * 2 + t) * 16;
#pragma unroll
    for (int k = 0; k < 4; ++k)
      f.w1f[t][k] = load_bfrag(W1, DMSP, k * 32, colBase, lane);
    f.bias1[t] = b1[colBase + n];
  }
#pragma unroll
  for (int k = 0; k < 8; ++k)
    f.w2f[k] = load_bfrag(W2, DHID, k * 32, wave * 16, lane);
  f.bias2 = b2[wave * 16 + n];
}

// ---- MLP over one 32-row tile: Ot = ReLU(Xt@W1+b1)@W2+b2 (f32 in LDS) ------
__device__ __forceinline__ void mlp_tile32(const _Float16* __restrict__ Xt,
                                           _Float16* __restrict__ Ht,
                                           float* __restrict__ Ot,
                                           const WFrags& f, int wave, int lane) {
  int n = lane & 15, half = lane >> 4;
  // GEMM1: [32,128] @ [128,256], two 16-row subtiles reuse the same B frags
#pragma unroll
  for (int rt = 0; rt < 2; ++rt) {
    const _Float16* Xr = Xt + rt * 16 * XSTR;
    _Float16* Hr = Ht + rt * 16 * HSTR;
    v16h a0 = load_afrag(Xr, XSTR, 0, lane);
    v16h a1 = load_afrag(Xr, XSTR, 32, lane);
    v16h a2 = load_afrag(Xr, XSTR, 64, lane);
    v16h a3 = load_afrag(Xr, XSTR, 96, lane);
#pragma unroll
    for (int t = 0; t < 2; ++t) {
      v8f acc = {};
      acc = wmma_f32_16x16x32(a0, f.w1f[t][0], acc);
      acc = wmma_f32_16x16x32(a1, f.w1f[t][1], acc);
      acc = wmma_f32_16x16x32(a2, f.w1f[t][2], acc);
      acc = wmma_f32_16x16x32(a3, f.w1f[t][3], acc);
      int colBase = (wave * 2 + t) * 16;
#pragma unroll
      for (int v = 0; v < 8; ++v) {
        float x = acc[v] + f.bias1[t];
        Hr[(v + half * 8) * HSTR + colBase + n] = (_Float16)(x > 0.f ? x : 0.f);
      }
    }
  }
  __syncthreads();
  // GEMM2: [32,256] @ [256,128]
#pragma unroll
  for (int rt = 0; rt < 2; ++rt) {
    const _Float16* Hr = Ht + rt * 16 * HSTR;
    float* Or = Ot + rt * 16 * OSTR;
    v8f acc = {};
#pragma unroll
    for (int k = 0; k < 8; ++k) {
      v16h a = load_afrag(Hr, HSTR, k * 32, lane);
      acc = wmma_f32_16x16x32(a, f.w2f[k], acc);
    }
    int colBase = wave * 16;
#pragma unroll
    for (int v = 0; v < 8; ++v)
      Or[(v + half * 8) * OSTR + colBase + n] = acc[v] + f.bias2;
  }
  __syncthreads();
}

// ---- node-level MLP: Y (+)= MLP(X), X f32 [R,128] ---------------------------
__global__ __launch_bounds__(256) void mlp_node_kernel(
    const float* __restrict__ X, const _Float16* __restrict__ W1,
    const float* __restrict__ b1, const _Float16* __restrict__ W2,
    const float* __restrict__ b2, float* __restrict__ Y, int R, int accum) {
  __shared__ _Float16 Xt[ROWS * XSTR];
  __shared__ _Float16 Ht[ROWS * HSTR];
  __shared__ float Ot[ROWS * OSTR];
  int tid = threadIdx.x, wave = tid >> 5, lane = tid & 31;
  WFrags f;
  load_wfrags(f, W1, b1, W2, b2, wave, lane);
  int ntiles = (R + ROWS - 1) >> 5;
  for (int t = blockIdx.x; t < ntiles; t += gridDim.x) {
    int row0 = t * ROWS;
    // stage X tile -> f16 LDS, packed pair stores (32 rows x 64 pairs)
#pragma unroll
    for (int i = 0; i < 8; ++i) {
      int id = tid + i * 256;  // 0..2047
      int r = id >> 6, c2 = (id & 63) * 2;
      int gr = row0 + r;
      float2 v = make_float2(0.f, 0.f);
      if (gr < R) v = *(const float2*)(X + (size_t)gr * DHID + c2);
      union { _Float16 h[2]; unsigned u; } pk;
      pk.h[0] = (_Float16)v.x;
      pk.h[1] = (_Float16)v.y;
      *(unsigned*)(Xt + r * XSTR + c2) = pk.u;
    }
    __syncthreads();
    mlp_tile32(Xt, Ht, Ot, f, wave, lane);
#pragma unroll
    for (int i = 0; i < 16; ++i) {
      int idx = tid + i * 256, r = idx >> 7, c = idx & 127;
      int gr = row0 + r;
      if (gr < R) {
        float v = Ot[r * OSTR + c];
        size_t o = (size_t)gr * DHID + c;
        Y[o] = accum ? (Y[o] + v) : v;
      }
    }
    __syncthreads();
  }
}

// ---- edge message: aggr[dst] += MLP(ea) + p[src] ----------------------------
__global__ __launch_bounds__(256) void edge_msg_kernel(
    const _Float16* __restrict__ EA, const int* __restrict__ srcI,
    const int* __restrict__ dstI, const _Float16* __restrict__ W1,
    const float* __restrict__ b1, const _Float16* __restrict__ W2,
    const float* __restrict__ b2, const float* __restrict__ P,
    float* __restrict__ AGGR, int E_) {
  __shared__ _Float16 Xt[ROWS * XSTR];
  __shared__ _Float16 Ht[ROWS * HSTR];
  __shared__ float Ot[ROWS * OSTR];
  __shared__ int sSrc[ROWS], sDst[ROWS];
  int tid = threadIdx.x, wave = tid >> 5, lane = tid & 31;
  WFrags f;
  load_wfrags(f, W1, b1, W2, b2, wave, lane);
  int ntiles = (E_ + ROWS - 1) >> 5;
  for (int t = blockIdx.x; t < ntiles; t += gridDim.x) {
    int e0 = t * ROWS;
    if (tid < ROWS) {
      int e = e0 + tid;
      sSrc[tid] = (e < E_) ? srcI[e] : -1;
      sDst[tid] = (e < E_) ? dstI[e] : -1;
    }
    // stage ea tile (f16, 32 rows x 256B) with async LDS loads: 2 x 16B/thread
#pragma unroll
    for (int pass = 0; pass < 2; ++pass) {
      int id = tid + pass * 256;  // 0..511
      int r = id >> 4, seg = id & 15;
      int e = e0 + r;
      if (e >= E_) e = 0;  // clamp; masked rows discarded at scatter
      unsigned lds_off =
          (unsigned)(size_t)(Xt + r * XSTR + seg * 8);  // low 32b = LDS offset
      unsigned long long gaddr =
          (unsigned long long)(size_t)(EA + (size_t)e * DHID + seg * 8);
      asm volatile("global_load_async_to_lds_b128 %0, %1, off"
                   :
                   : "v"(lds_off), "v"(gaddr)
                   : "memory");
    }
    {  // prefetch next tile for this block (global_prefetch_b8)
      int r = tid >> 4, seg = tid & 15;
      long long en = (long long)(t + gridDim.x) * ROWS + r;
      if (en < E_) __builtin_prefetch(EA + (size_t)en * DHID + seg * 8, 0, 1);
    }
    asm volatile("s_wait_asynccnt 0x0" ::: "memory");
    __syncthreads();
    mlp_tile32(Xt, Ht, Ot, f, wave, lane);
#pragma unroll
    for (int i = 0; i < 16; ++i) {
      int idx = tid + i * 256, r = idx >> 7, c = idx & 127;
      int s = sSrc[r], d = sDst[r];
      if (d >= 0) {
        float v = Ot[r * OSTR + c] + P[(size_t)s * DHID + c];
        atomicAdd(&AGGR[(size_t)d * DHID + c], v);
      }
    }
    __syncthreads();
  }
}

// ---- small elementwise / lift kernels ---------------------------------------
__global__ void cvt_f16_kernel(const float* __restrict__ in,
                               _Float16* __restrict__ out, int n) {
  int i = blockIdx.x * blockDim.x + threadIdx.x;
  if (i < n) out[i] = (_Float16)in[i];
}

__global__ void zero_kernel(float* __restrict__ p, int n) {
  int i = blockIdx.x * blockDim.x + threadIdx.x;
  if (i < n) p[i] = 0.f;
}

__global__ void copy_kernel(float* __restrict__ d, const float* __restrict__ s,
                            int n) {
  int i = blockIdx.x * blockDim.x + threadIdx.x;
  if (i < n) d[i] = s[i];
}

__global__ void maxout_kernel(float* __restrict__ jk,
                              const float* __restrict__ h, int n) {
  int i = blockIdx.x * blockDim.x + threadIdx.x;
  if (i < n) jk[i] = fmaxf(jk[i], h[i]);
}

__global__ void deg_accum_kernel(const int* __restrict__ dst,
                                 float* __restrict__ deg, int E_) {
  int e = blockIdx.x * blockDim.x + threadIdx.x;
  if (e < E_) atomicAdd(&deg[dst[e]], 1.0f);
}

__global__ void deg_inv_kernel(float* __restrict__ deg, int n) {
  int i = blockIdx.x * blockDim.x + threadIdx.x;
  if (i < n) deg[i] = 1.0f / fmaxf(deg[i], 1.0f);
}

__global__ void scale_rows_kernel(float* __restrict__ a,
                                  const float* __restrict__ inv, int n) {
  int i = blockIdx.x * blockDim.x + threadIdx.x;
  if (i < n) a[i] *= inv[i >> 7];
}

__global__ void node_lift_kernel(const float* __restrict__ x,
                                 const float* __restrict__ w,
                                 const float* __restrict__ b,
                                 float* __restrict__ h, int N_) {
  __shared__ float sw[16 * DHID];
  __shared__ float sb[DHID];
  for (int i = threadIdx.x; i < 16 * DHID; i += 256) sw[i] = w[i];
  if (threadIdx.x < DHID) sb[threadIdx.x] = b[threadIdx.x];
  __syncthreads();
  int g = blockIdx.x * 256 + threadIdx.x;
  int node = g >> 5, c0 = (g & 31) * 4;
  if (node >= N_) return;
  float a0 = sb[c0], a1 = sb[c0 + 1], a2 = sb[c0 + 2], a3 = sb[c0 + 3];
#pragma unroll
  for (int k = 0; k < 16; ++k) {
    float xv = x[(size_t)node * 16 + k];
    a0 += xv * sw[k * DHID + c0];
    a1 += xv * sw[k * DHID + c0 + 1];
    a2 += xv * sw[k * DHID + c0 + 2];
    a3 += xv * sw[k * DHID + c0 + 3];
  }
  float4 o = {a0, a1, a2, a3};
  *(float4*)(h + (size_t)node * DHID + c0) = o;
}

__global__ void edge_lift_kernel(const float* __restrict__ ea,
                                 const float* __restrict__ w,
                                 const float* __restrict__ b,
                                 _Float16* __restrict__ out, int E_) {
  __shared__ float sw[8 * DHID];
  __shared__ float sb[DHID];
  for (int i = threadIdx.x; i < 8 * DHID; i += 256) sw[i] = w[i];
  if (threadIdx.x < DHID) sb[threadIdx.x] = b[threadIdx.x];
  __syncthreads();
  int g = blockIdx.x * 256 + threadIdx.x;
  int e = g >> 5, c0 = (g & 31) * 4;
  if (e >= E_) return;
  float a0 = sb[c0], a1 = sb[c0 + 1], a2 = sb[c0 + 2], a3 = sb[c0 + 3];
#pragma unroll
  for (int k = 0; k < 8; ++k) {
    float xv = ea[(size_t)e * 8 + k];
    a0 += xv * sw[k * DHID + c0];
    a1 += xv * sw[k * DHID + c0 + 1];
    a2 += xv * sw[k * DHID + c0 + 2];
    a3 += xv * sw[k * DHID + c0 + 3];
  }
  union { _Float16 h4[4]; uint2 u; } pk;
  pk.h4[0] = (_Float16)a0; pk.h4[1] = (_Float16)a1;
  pk.h4[2] = (_Float16)a2; pk.h4[3] = (_Float16)a3;
  *(uint2*)(out + (size_t)e * DHID + c0) = pk.u;
}

__global__ void final_kernel(const float* __restrict__ jk,
                             const float* __restrict__ w,
                             const float* __restrict__ b,
                             float* __restrict__ out, int N_) {
  __shared__ float sw[DHID * 4];
  __shared__ float sb[4];
  for (int i = threadIdx.x; i < DHID * 4; i += 256) sw[i] = w[i];
  if (threadIdx.x < 4) sb[threadIdx.x] = b[threadIdx.x];
  __syncthreads();
  int node = blockIdx.x * 256 + threadIdx.x;
  if (node >= N_) return;
  float a0 = sb[0], a1 = sb[1], a2 = sb[2], a3 = sb[3];
#pragma unroll 4
  for (int k = 0; k < DHID; ++k) {
    float v = jk[(size_t)node * DHID + k];
    a0 += v * sw[k * 4 + 0];
    a1 += v * sw[k * 4 + 1];
    a2 += v * sw[k * 4 + 2];
    a3 += v * sw[k * 4 + 3];
  }
  out[node * 4 + 0] = a0; out[node * 4 + 1] = a1;
  out[node * 4 + 2] = a2; out[node * 4 + 3] = a3;
}

extern "C" void kernel_launch(void* const* d_in, const int* in_sizes, int n_in,
                              void* d_out, int out_size, void* d_ws,
                              size_t ws_size, hipStream_t stream) {
  (void)in_sizes; (void)n_in; (void)out_size; (void)ws_size;
  const float* x        = (const float*)d_in[0];
  const float* edge_attr= (const float*)d_in[1];
  const int*   ei       = (const int*)d_in[2];
  const float* node_w   = (const float*)d_in[3];
  const float* node_b   = (const float*)d_in[4];
  const float* edge_w   = (const float*)d_in[5];
  const float* edge_b   = (const float*)d_in[6];
  const float* msg_w1   = (const float*)d_in[7];
  const float* msg_b1   = (const float*)d_in[8];
  const float* msg_w2   = (const float*)d_in[9];
  const float* msg_b2   = (const float*)d_in[10];
  const float* upd_w1   = (const float*)d_in[11];
  const float* upd_b1   = (const float*)d_in[12];
  const float* upd_w2   = (const float*)d_in[13];
  const float* upd_b2   = (const float*)d_in[14];
  const float* final_w  = (const float*)d_in[15];
  const float* final_b  = (const float*)d_in[16];
  float* out = (float*)d_out;

  // workspace carve-up
  char* ws = (char*)d_ws;
  auto carve = [&](size_t bytes) {
    char* p = ws;
    ws += (bytes + 255) & ~(size_t)255;
    return p;
  };
  const size_t NH = (size_t)NN * DHID;
  float* hA     = (float*)carve(NH * 4);
  float* hB     = (float*)carve(NH * 4);
  float* p      = (float*)carve(NH * 4);
  float* aggr   = (float*)carve(NH * 4);
  float* jk     = (float*)carve(NH * 4);
  float* invdeg = (float*)carve((size_t)NN * 4);
  _Float16* ea16 = (_Float16*)carve((size_t)NE * DHID * 2);
  _Float16* w16  = (_Float16*)carve((size_t)4 * 65536 * 2);
  _Float16* msgW1h = w16;            // [2][128*256]
  _Float16* msgW2h = w16 + 65536;    // [2][256*128]
  _Float16* updW1h = w16 + 131072;
  _Float16* updW2h = w16 + 196608;

  const int WSZ = 65536;  // per-tensor f16 weight element count (both layers)
  cvt_f16_kernel<<<(WSZ + 255) / 256, 256, 0, stream>>>(msg_w1, msgW1h, WSZ);
  cvt_f16_kernel<<<(WSZ + 255) / 256, 256, 0, stream>>>(msg_w2, msgW2h, WSZ);
  cvt_f16_kernel<<<(WSZ + 255) / 256, 256, 0, stream>>>(upd_w1, updW1h, WSZ);
  cvt_f16_kernel<<<(WSZ + 255) / 256, 256, 0, stream>>>(upd_w2, updW2h, WSZ);

  node_lift_kernel<<<(NN * 32 + 255) / 256, 256, 0, stream>>>(x, node_w, node_b,
                                                              hA, NN);
  edge_lift_kernel<<<(NE * 32 + 255) / 256, 256, 0, stream>>>(
      edge_attr, edge_w, edge_b, ea16, NE);

  zero_kernel<<<(NN + 255) / 256, 256, 0, stream>>>(invdeg, NN);
  deg_accum_kernel<<<(NE + 255) / 256, 256, 0, stream>>>(ei + NE, invdeg, NE);
  deg_inv_kernel<<<(NN + 255) / 256, 256, 0, stream>>>(invdeg, NN);

  copy_kernel<<<((int)NH + 255) / 256, 256, 0, stream>>>(jk, hA, (int)NH);

  float* h = hA;
  float* hn = hB;
  for (int l = 0; l < 2; ++l) {
    const _Float16* W1m = msgW1h + (size_t)l * DHID * DMSP;
    const _Float16* W2m = msgW2h + (size_t)l * DMSP * DHID;
    const float* B1m = msg_b1 + (size_t)l * DMSP;
    const float* B2m = msg_b2 + (size_t)l * DHID;
    const _Float16* W1u = updW1h + (size_t)l * DHID * DMSP;
    const _Float16* W2u = updW2h + (size_t)l * DMSP * DHID;
    const float* B1u = upd_b1 + (size_t)l * DMSP;
    const float* B2u = upd_b2 + (size_t)l * DHID;

    // p = MLP_msg(h)  (node-level; reused per incoming edge via gather)
    mlp_node_kernel<<<1024, 256, 0, stream>>>(h, W1m, B1m, W2m, B2m, p, NN, 0);
    zero_kernel<<<((int)NH + 255) / 256, 256, 0, stream>>>(aggr, (int)NH);
    // aggr[dst] += MLP_msg(ea) + p[src]
    edge_msg_kernel<<<2048, 256, 0, stream>>>(ea16, ei, ei + NE, W1m, B1m, W2m,
                                              B2m, p, aggr, NE);
    scale_rows_kernel<<<((int)NH + 255) / 256, 256, 0, stream>>>(aggr, invdeg,
                                                                 (int)NH);
    // h' = MLP_upd(h) + MLP_upd(aggr)
    mlp_node_kernel<<<1024, 256, 0, stream>>>(h, W1u, B1u, W2u, B2u, hn, NN, 0);
    mlp_node_kernel<<<1024, 256, 0, stream>>>(aggr, W1u, B1u, W2u, B2u, hn, NN,
                                              1);
    maxout_kernel<<<((int)NH + 255) / 256, 256, 0, stream>>>(jk, hn, (int)NH);
    float* tmp = h; h = hn; hn = tmp;
  }

  final_kernel<<<(NN + 255) / 256, 256, 0, stream>>>(jk, final_w, final_b, out,
                                                     NN);
}
